// AllegroGDML_Module_80247168958613
// MI455X (gfx1250) — matrix operand order repaired and checked by reading, not compiled, not run on hardware
//
#include <hip/hip_runtime.h>
#include <hip/hip_bf16.h>
#include <stdint.h>

// ---------------------------------------------------------------------------
// CDNA5 (gfx1250) bf16 WMMA helpers
// ---------------------------------------------------------------------------
typedef __attribute__((ext_vector_type(16))) __bf16 v16bf;
typedef __attribute__((ext_vector_type(8)))  float  v8f;

__device__ __forceinline__ unsigned short f2bf(float f) {
  unsigned int u = __float_as_uint(f);
  unsigned int r = (u + 0x7FFFu + ((u >> 16) & 1u)) >> 16;  // RNE
  return (unsigned short)r;
}

__device__ __forceinline__ v8f wmma_bf16(v16bf a, v16bf b, v8f c) {
  return __builtin_amdgcn_wmma_f32_16x16x32_bf16(false, a, false, b, (short)0, c,
                                                 false, false);
}

// A fragment: 16(M) x 32(K) bf16 tile from LDS row-major [16][stride].
// lane<16 holds row M=lane, K {0..15}; lane>=16 K {16..31} (matches B swizzle).
__device__ __forceinline__ v16bf load_a_frag(const unsigned short* A, int stride,
                                             int kc, int lane) {
  const unsigned short* row = A + (lane & 15) * stride + kc * 32 + ((lane >> 4) << 4);
  union { v16bf v; uint4 q[2]; } a;
  const uint4* q = (const uint4*)row;
  a.q[0] = q[0];   // K base + 0..7
  a.q[1] = q[1];   // K base + 8..15
  return a.v;
}

// B fragment from pre-swizzled weights: element ((kc*NT+nt)*32+lane)*16 + j
// lane holds column N = lane&15; K = kc*32 + (lane>=16?16:0) + j.
__device__ __forceinline__ v16bf load_b_frag(const unsigned short* B, int NT,
                                             int kc, int nt, int lane) {
  const unsigned short* p = B + (((size_t)(kc * NT + nt) * 32 + lane) << 4);
  union { v16bf v; uint4 q[2]; } b;
  b.q[0] = *(const uint4*)p;
  b.q[1] = *(const uint4*)(p + 8);
  return b.v;
}

// Full-width 16xK @ KxN (N = NT*16) with register-resident accumulators.
template <int NT>
__device__ __forceinline__ void gemm_acc(const unsigned short* A, int stride, int K,
                                         const unsigned short* B, v8f (&acc)[NT],
                                         int lane) {
  #pragma unroll
  for (int nt = 0; nt < NT; ++nt) acc[nt] = 0.0f;
  for (int kc = 0; kc < (K >> 5); ++kc) {
    v16bf a = load_a_frag(A, stride, kc, lane);
    #pragma unroll
    for (int nt = 0; nt < NT; ++nt) {
      v16bf b = load_b_frag(B, NT, kc, nt, lane);
      acc[nt] = wmma_bf16(a, b, acc[nt]);
    }
  }
}

// Apply SiLU to C fragments and store bf16 into LDS row-major [16][stride].
// C layout (ISA): lane holds col N=lane&15, rows (lane>=16?8:0)+r.
template <int NT>
__device__ __forceinline__ void store_silu_bf16(unsigned short* O, int stride,
                                                const v8f (&acc)[NT], int lane) {
  int col = lane & 15, rb = (lane >> 4) << 3;
  #pragma unroll
  for (int nt = 0; nt < NT; ++nt)
    #pragma unroll
    for (int r = 0; r < 8; ++r) {
      float x = acc[nt][r];
      O[(rb + r) * stride + nt * 16 + col] = f2bf(x / (1.f + __expf(-x)));
    }
}

// ---------------------------------------------------------------------------
// Weight pre-pass: f32 [Ksrc][N] -> bf16 B-fragment-contiguous, K padded to 32.
// ---------------------------------------------------------------------------
__global__ void swizzle_weight(const float* __restrict__ src,
                               unsigned short* __restrict__ dst,
                               int N, int Ksrc, int total) {
  int idx = blockIdx.x * blockDim.x + threadIdx.x;
  if (idx >= total) return;
  int j = idx & 15;
  int lane = (idx >> 4) & 31;
  int frag = idx >> 9;
  int NT = N >> 4;
  int nt = frag % NT;
  int kc = frag / NT;
  int k = kc * 32 + ((lane >> 4) << 4) + j;
  int n = nt * 16 + (lane & 15);
  float v = (k < Ksrc) ? src[(size_t)k * N + n] : 0.f;
  dst[idx] = f2bf(v);
}

// ---------------------------------------------------------------------------
// Phase 0: x2b -> lat (w2b MLP, cut), env0 MLP -> w_edge (fs/fv) + w_env scatter
// ---------------------------------------------------------------------------
#define P0_WAVES 4
__global__ __launch_bounds__(128) void edge_phase0(
    const float* __restrict__ edge_attr, const float* __restrict__ node_attrs,
    const float* __restrict__ edge_embed, const float* __restrict__ edge_u,
    const int* __restrict__ eidx,
    const unsigned short* __restrict__ w2b1s, const unsigned short* __restrict__ w2b2s,
    const unsigned short* __restrict__ env0_1s, const unsigned short* __restrict__ env0_2s,
    float* __restrict__ lat_g, float* __restrict__ fs_g, float* __restrict__ fv_g,
    float* __restrict__ cut_g, float* __restrict__ env_s, float* __restrict__ env_v,
    int E) {
  __shared__ unsigned short inbf[P0_WAVES][16 * 32];
  __shared__ unsigned short hbf[P0_WAVES][16 * 128];
  __shared__ float cutsh[P0_WAVES][16];
  __shared__ float attrsh[P0_WAVES][16][4];
  __shared__ int cidxsh[P0_WAVES][16];

  const int lane = threadIdx.x & 31;
  const int w = threadIdx.x >> 5;
  const int e0 = (blockIdx.x * P0_WAVES + w) * 16;
  const int colb = lane & 15, rb = (lane >> 4) << 3;

  if (lane < 16) {
    int e = e0 + lane;
    unsigned short* row = &inbf[w][lane * 32];
    if (e < E) {
      int c = eidx[e], n = eidx[E + e];
      cidxsh[w][lane] = c;
      #pragma unroll
      for (int t = 0; t < 4; ++t) {
        row[t]     = f2bf(node_attrs[(size_t)c * 4 + t]);
        row[4 + t] = f2bf(node_attrs[(size_t)n * 4 + t]);
        attrsh[w][lane][t] = edge_attr[(size_t)e * 4 + t];
      }
      #pragma unroll
      for (int t = 0; t < 8; ++t) row[8 + t] = f2bf(edge_embed[(size_t)e * 8 + t]);
      float u = edge_u[e];
      float u2 = u * u, u6 = u2 * u2 * u2;
      float f = 1.f - 28.f * u6 + 48.f * u6 * u - 21.f * u6 * u2;  // p=6 cutoff
      float ct = (u < 1.f) ? f : 0.f;
      cutsh[w][lane] = ct;
      cut_g[e] = ct;
    } else {
      cidxsh[w][lane] = 0;
      cutsh[w][lane] = 0.f;
      #pragma unroll
      for (int t = 0; t < 4; ++t) attrsh[w][lane][t] = 0.f;
      #pragma unroll
      for (int t = 0; t < 16; ++t) row[t] = 0;
    }
    #pragma unroll
    for (int t = 16; t < 32; ++t) row[t] = 0;  // K pad 16->32
  }
  __syncthreads();

  {  // h = silu(x2b @ w2b1)  (K=32 padded)
    v8f h[8];
    gemm_acc<8>(&inbf[w][0], 32, 32, w2b1s, h, lane);
    store_silu_bf16<8>(&hbf[w][0], 128, h, lane);
  }
  __syncthreads();
  {  // lat = cut * (h @ w2b2)
    v8f o[8];
    gemm_acc<8>(&hbf[w][0], 128, 128, w2b2s, o, lane);
    __syncthreads();
    #pragma unroll
    for (int nt = 0; nt < 8; ++nt)
      #pragma unroll
      for (int r = 0; r < 8; ++r) {
        int m = rb + r, c = nt * 16 + colb, e = e0 + m;
        float v = cutsh[w][m] * o[nt][r];
        if (e < E) lat_g[(size_t)e * 128 + c] = v;
        hbf[w][m * 128 + c] = f2bf(v);
      }
  }
  __syncthreads();
  {  // h = silu(lat @ env0_w1)
    v8f h[8];
    gemm_acc<8>(&hbf[w][0], 128, 128, env0_1s, h, lane);
    __syncthreads();
    store_silu_bf16<8>(&hbf[w][0], 128, h, lane);
  }
  __syncthreads();
  {  // w0 = h @ env0_w2 (64 cols): [0:32]=w_edge -> fs/fv, [32:64]=w_env -> scatter
    v8f o[4];
    gemm_acc<4>(&hbf[w][0], 128, 128, env0_2s, o, lane);
    const float norm = 0.223606798f;  // 1/sqrt(AVG_N=20)
    #pragma unroll
    for (int nt = 0; nt < 4; ++nt)
      #pragma unroll
      for (int r = 0; r < 8; ++r) {
        int m = rb + r, e = e0 + m;
        if (e >= E) continue;
        int c = nt * 16 + colb;
        float val = o[nt][r];
        float a0 = attrsh[w][m][0];
        if (c < 32) {
          int u = c >> 1;
          if ((c & 1) == 0) {
            fs_g[(size_t)e * 16 + u] = val * a0;
          } else {
            #pragma unroll
            for (int i = 0; i < 3; ++i)
              fv_g[((size_t)e * 16 + u) * 3 + i] = val * attrsh[w][m][1 + i];
          }
        } else {
          int cc = c - 32, u = cc >> 1;
          int cen = cidxsh[w][m];
          if ((cc & 1) == 0) {
            atomicAdd(&env_s[(size_t)cen * 16 + u], norm * val * a0);
          } else {
            #pragma unroll
            for (int i = 0; i < 3; ++i)
              atomicAdd(&env_v[((size_t)cen * 16 + u) * 3 + i],
                        norm * val * attrsh[w][m][1 + i]);
          }
        }
      }
  }
}

// ---------------------------------------------------------------------------
// Per-node envlin mixing (tiny 16x16 per node)
// ---------------------------------------------------------------------------
__global__ void node_linear(const float* __restrict__ acc_s, const float* __restrict__ acc_v,
                            const float* __restrict__ ws, const float* __restrict__ wv,
                            float* __restrict__ lin_s, float* __restrict__ lin_v, int Nn) {
  int idx = blockIdx.x * blockDim.x + threadIdx.x;
  if (idx >= Nn * 16) return;
  int n = idx >> 4, v = idx & 15;
  float s = 0.f, vx = 0.f, vy = 0.f, vz = 0.f;
  for (int u = 0; u < 16; ++u) {
    float w1 = ws[u * 16 + v], w2 = wv[u * 16 + v];
    s  += acc_s[(size_t)n * 16 + u] * w1;
    vx += acc_v[((size_t)n * 16 + u) * 3 + 0] * w2;
    vy += acc_v[((size_t)n * 16 + u) * 3 + 1] * w2;
    vz += acc_v[((size_t)n * 16 + u) * 3 + 2] * w2;
  }
  const float inv_sm = 0.25f;  // 1/sqrt(MUL=16)
  lin_s[idx] = s * inv_sm;
  lin_v[(size_t)idx * 3 + 0] = vx * inv_sm;
  lin_v[(size_t)idx * 3 + 1] = vy * inv_sm;
  lin_v[(size_t)idx * 3 + 2] = vz * inv_sm;
}

// ---------------------------------------------------------------------------
// Layer kernel: gather env, build S/V + latent_in, streamed l2w GEMM with
// in-register S/V contraction; layer0 fuses lat1 residual + env1 scatter;
// layer1 fuses fl2w + final dot -> out.
// ---------------------------------------------------------------------------
#define L_WAVES 2
template <int LAYER>
__global__ __launch_bounds__(64) void edge_layer(
    const float* __restrict__ edge_attr, const int* __restrict__ eidx,
    float* __restrict__ lat_g, float* __restrict__ fs_g, float* __restrict__ fv_g,
    const float* __restrict__ cut_g,
    const float* __restrict__ env_s_lin, const float* __restrict__ env_v_lin,
    float* __restrict__ env_s_acc, float* __restrict__ env_v_acc,
    const unsigned short* __restrict__ l2w_1s, const unsigned short* __restrict__ l2w_2s,
    const unsigned short* __restrict__ lat1_1s, const unsigned short* __restrict__ lat1_2s,
    const unsigned short* __restrict__ env1_1s, const unsigned short* __restrict__ env1_2s,
    const unsigned short* __restrict__ fl2w_1s, const unsigned short* __restrict__ fl2w_2s,
    float* __restrict__ out, int E) {
  __shared__ unsigned short inbf[L_WAVES][16 * 160];
  __shared__ unsigned short hbf[L_WAVES][16 * 128];
  __shared__ float Ssh[L_WAVES][16 * 32];    // S[p,u] per edge row
  __shared__ float Vsh[L_WAVES][16 * 144];   // V[p,u,i] per edge row
  __shared__ float cutsh[L_WAVES][16];
  __shared__ float attrsh[L_WAVES][16][4];
  __shared__ int cidxsh[L_WAVES][16];

  const int lane = threadIdx.x & 31;
  const int w = threadIdx.x >> 5;
  const int e0 = (blockIdx.x * L_WAVES + w) * 16;
  const int colb = lane & 15, rb = (lane >> 4) << 3;
  const float inv_s3 = 0.577350269f, inv_s2 = 0.707106781f;

  if (lane < 16) {
    int e = e0 + lane;
    bool ok = e < E;
    int cen = ok ? eidx[e] : 0;
    cidxsh[w][lane] = cen;
    cutsh[w][lane] = (LAYER == 0 && ok) ? cut_g[e] : 0.f;
    unsigned short* row = &inbf[w][lane * 160];
    #pragma unroll
    for (int t = 0; t < 4; ++t) attrsh[w][lane][t] = ok ? edge_attr[(size_t)e * 4 + t] : 0.f;
    for (int k = 0; k < 128; ++k)
      row[k] = ok ? f2bf(lat_g[(size_t)e * 128 + k]) : (unsigned short)0;
    for (int u = 0; u < 16; ++u) {
      float fsv = ok ? fs_g[(size_t)e * 16 + u] : 0.f;
      float es  = ok ? env_s_lin[(size_t)cen * 16 + u] : 0.f;
      float f0 = ok ? fv_g[((size_t)e * 16 + u) * 3 + 0] : 0.f;
      float f1 = ok ? fv_g[((size_t)e * 16 + u) * 3 + 1] : 0.f;
      float f2 = ok ? fv_g[((size_t)e * 16 + u) * 3 + 2] : 0.f;
      float g0 = ok ? env_v_lin[((size_t)cen * 16 + u) * 3 + 0] : 0.f;
      float g1 = ok ? env_v_lin[((size_t)cen * 16 + u) * 3 + 1] : 0.f;
      float g2 = ok ? env_v_lin[((size_t)cen * 16 + u) * 3 + 2] : 0.f;
      float s1 = fsv * es;
      float s2 = (f0 * g0 + f1 * g1 + f2 * g2) * inv_s3;
      Ssh[w][lane * 32 + u] = s1;
      Ssh[w][lane * 32 + 16 + u] = s2;
      float* Vr = &Vsh[w][lane * 144];
      Vr[u * 3 + 0] = fsv * g0;  Vr[u * 3 + 1] = fsv * g1;  Vr[u * 3 + 2] = fsv * g2;  // v1
      Vr[(16 + u) * 3 + 0] = f0 * es; Vr[(16 + u) * 3 + 1] = f1 * es; Vr[(16 + u) * 3 + 2] = f2 * es;  // v2
      Vr[(32 + u) * 3 + 0] = (f1 * g2 - f2 * g1) * inv_s2;  // v3 = cross
      Vr[(32 + u) * 3 + 1] = (f2 * g0 - f0 * g2) * inv_s2;
      Vr[(32 + u) * 3 + 2] = (f0 * g1 - f1 * g0) * inv_s2;
      row[128 + 2 * u]     = f2bf(s1);   // interleaved scalars
      row[128 + 2 * u + 1] = f2bf(s2);
    }
  }
  __syncthreads();

  {  // l2w hidden: silu(latent_in @ lw1), K=160
    v8f h[8];
    gemm_acc<8>(&inbf[w][0], 160, 160, l2w_1s, h, lane);
    store_silu_bf16<8>(&hbf[w][0], 128, h, lane);
  }
  __syncthreads();

  float fsacc[8];
  float fvacc[8][3];
  #pragma unroll
  for (int r = 0; r < 8; ++r) {
    fsacc[r] = 0.f;
    fvacc[r][0] = fvacc[r][1] = fvacc[r][2] = 0.f;
  }
  {  // streamed 128 -> 1280 GEMM, tile nt == (p*16+u); consume immediately
    v16bf afr[4];
    #pragma unroll
    for (int kc = 0; kc < 4; ++kc) afr[kc] = load_a_frag(&hbf[w][0], 128, kc, lane);
    const int nt0 = (LAYER == 0) ? 0 : 32;  // layer1: fs is dead, skip ws half
    for (int nt = nt0; nt < 80; ++nt) {
      v8f c = 0.0f;
      #pragma unroll
      for (int kc = 0; kc < 4; ++kc) {
        v16bf b = load_b_frag(l2w_2s, 80, kc, nt, lane);
        c = wmma_bf16(afr[kc], b, c);
      }
      if (nt < 32) {
        #pragma unroll
        for (int r = 0; r < 8; ++r)
          fsacc[r] += Ssh[w][(rb + r) * 32 + nt] * c[r];
      } else {
        int pu = nt - 32;
        #pragma unroll
        for (int r = 0; r < 8; ++r) {
          float sc = c[r];
          const float* Vr = &Vsh[w][(rb + r) * 144 + pu * 3];
          fvacc[r][0] += Vr[0] * sc;
          fvacc[r][1] += Vr[1] * sc;
          fvacc[r][2] += Vr[2] * sc;
        }
      }
    }
  }

  if (LAYER == 0) {
    const float sA = 0.176776695f;  // 1/sqrt(2*MUL)
    const float sB = 0.144337567f;  // 1/sqrt(3*MUL)
    #pragma unroll
    for (int r = 0; r < 8; ++r) {
      int e = e0 + rb + r;
      if (e < E) {
        fs_g[(size_t)e * 16 + colb] = fsacc[r] * sA;
        #pragma unroll
        for (int i = 0; i < 3; ++i)
          fv_g[((size_t)e * 16 + colb) * 3 + i] = fvacc[r][i] * sB;
      }
    }
    __syncthreads();
    {  // lat1 hidden
      v8f h[8];
      gemm_acc<8>(&inbf[w][0], 160, 160, lat1_1s, h, lane);
      __syncthreads();
      store_silu_bf16<8>(&hbf[w][0], 128, h, lane);
    }
    __syncthreads();
    {  // residual lat update: lat = c1*lat + 0.5*c1*cut*(h @ lat1_w2)
      v8f o[8];
      gemm_acc<8>(&hbf[w][0], 128, 128, lat1_2s, o, lane);
      const float c1 = 0.894427191f, ac1 = 0.447213595f;  // 1/sqrt(1.25), 0.5*that
      __syncthreads();
      #pragma unroll
      for (int nt = 0; nt < 8; ++nt)
        #pragma unroll
        for (int r = 0; r < 8; ++r) {
          int m = rb + r, c = nt * 16 + colb, e = e0 + m;
          unsigned short bv = 0;
          if (e < E) {
            float nv = c1 * lat_g[(size_t)e * 128 + c] + ac1 * cutsh[w][m] * o[nt][r];
            lat_g[(size_t)e * 128 + c] = nv;
            bv = f2bf(nv);
          }
          inbf[w][m * 160 + c] = bv;  // new lat as env1 input
        }
    }
    __syncthreads();
    {  // env1 hidden (input = new lat, stride 160, K=128)
      v8f h[8];
      gemm_acc<8>(&inbf[w][0], 160, 128, env1_1s, h, lane);
      __syncthreads();
      store_silu_bf16<8>(&hbf[w][0], 128, h, lane);
    }
    __syncthreads();
    {  // w_env = h @ env1_w2 (32 cols) -> weighter -> scatter atomics
      v8f o[2];
      gemm_acc<2>(&hbf[w][0], 128, 128, env1_2s, o, lane);
      const float norm = 0.223606798f;
      #pragma unroll
      for (int nt = 0; nt < 2; ++nt)
        #pragma unroll
        for (int r = 0; r < 8; ++r) {
          int m = rb + r, e = e0 + m;
          if (e >= E) continue;
          int c = nt * 16 + colb, u = c >> 1;
          int cen = cidxsh[w][m];
          float val = o[nt][r];
          if ((c & 1) == 0) {
            atomicAdd(&env_s_acc[(size_t)cen * 16 + u], norm * val * attrsh[w][m][0]);
          } else {
            #pragma unroll
            for (int i = 0; i < 3; ++i)
              atomicAdd(&env_v_acc[((size_t)cen * 16 + u) * 3 + i],
                        norm * val * attrsh[w][m][1 + i]);
          }
        }
    }
  } else {
    // final: fw = mlp(latent_in, fl2w); out = 0.25 * (fw . fv_new)
    const float sB = 0.144337567f;
    __syncthreads();
    #pragma unroll
    for (int r = 0; r < 8; ++r) {  // stage scaled fv_new into (free) V p=0 region
      float* Vr = &Vsh[w][(rb + r) * 144 + colb * 3];
      Vr[0] = fvacc[r][0] * sB;
      Vr[1] = fvacc[r][1] * sB;
      Vr[2] = fvacc[r][2] * sB;
    }
    {
      v8f h[8];
      gemm_acc<8>(&inbf[w][0], 160, 160, fl2w_1s, h, lane);
      __syncthreads();
      store_silu_bf16<8>(&hbf[w][0], 128, h, lane);
    }
    __syncthreads();
    {
      v8f o[1];
      gemm_acc<1>(&hbf[w][0], 128, 128, fl2w_2s, o, lane);
      #pragma unroll
      for (int r = 0; r < 8; ++r) Ssh[w][(rb + r) * 32 + colb] = o[0][r];  // fw
    }
    __syncthreads();
    if (lane < 16) {
      int e = e0 + lane;
      if (e < E) {
        #pragma unroll
        for (int i = 0; i < 3; ++i) {
          float acc = 0.f;
          for (int u = 0; u < 16; ++u)
            acc += Ssh[w][lane * 32 + u] * Vsh[w][lane * 144 + u * 3 + i];
          out[(size_t)e * 3 + i] = acc * 0.25f;
        }
      }
    }
  }
}

// ---------------------------------------------------------------------------
// Host orchestration
// ---------------------------------------------------------------------------
extern "C" void kernel_launch(void* const* d_in, const int* in_sizes, int n_in,
                              void* d_out, int out_size, void* d_ws, size_t ws_size,
                              hipStream_t stream) {
  (void)n_in; (void)out_size; (void)ws_size;
  const float* edge_attr  = (const float*)d_in[0];
  const float* node_attrs = (const float*)d_in[1];
  const float* edge_embed = (const float*)d_in[2];
  const float* edge_u     = (const float*)d_in[3];
  const int*   eidx       = (const int*)d_in[4];
  const float* envlin_ws  = (const float*)d_in[17];
  const float* envlin_wv  = (const float*)d_in[18];
  const int E  = in_sizes[0] / 4;
  const int Nn = in_sizes[1] / 4;

  // {input idx, Ksrc, Kpad, N}
  static const int spec[14][4] = {
      {5, 16, 32, 128},   {6, 128, 128, 128}, {7, 160, 160, 128}, {8, 128, 128, 128},
      {9, 128, 128, 128}, {10, 128, 128, 64}, {11, 128, 128, 128},{12, 128, 128, 32},
      {13, 160, 160, 128},{14, 128, 128, 1280},{15, 160, 160, 128},{16, 128, 128, 1280},
      {19, 160, 160, 128},{20, 128, 128, 16}};
  size_t woff[14], welems = 0;
  for (int i = 0; i < 14; ++i) { woff[i] = welems; welems += (size_t)spec[i][2] * spec[i][3]; }

  char* base = (char*)d_ws;
  size_t cur = 0;
  auto alloc = [&](size_t bytes) -> char* {
    char* p = base + cur;
    cur += (bytes + 255) & ~(size_t)255;
    return p;
  };
  unsigned short* wsw = (unsigned short*)alloc(welems * 2);
  float* lat    = (float*)alloc((size_t)E * 128 * 4);
  float* fs     = (float*)alloc((size_t)E * 16 * 4);
  float* fv     = (float*)alloc((size_t)E * 48 * 4);
  float* cut    = (float*)alloc((size_t)E * 4);
  float* envacc = (float*)alloc((size_t)Nn * 64 * 4);
  float* envlin = (float*)alloc((size_t)Nn * 64 * 4);
  float* env_s_acc = envacc;
  float* env_v_acc = envacc + (size_t)Nn * 16;
  float* env_s_lin = envlin;
  float* env_v_lin = envlin + (size_t)Nn * 16;

  for (int i = 0; i < 14; ++i) {
    int total = spec[i][2] * spec[i][3];
    swizzle_weight<<<dim3((total + 255) / 256), dim3(256), 0, stream>>>(
        (const float*)d_in[spec[i][0]], wsw + woff[i], spec[i][3], spec[i][1], total);
  }

  hipMemsetAsync(envacc, 0, (size_t)Nn * 64 * 4, stream);
  edge_phase0<<<dim3((E + 63) / 64), dim3(128), 0, stream>>>(
      edge_attr, node_attrs, edge_embed, edge_u, eidx,
      wsw + woff[0], wsw + woff[1], wsw + woff[4], wsw + woff[5],
      lat, fs, fv, cut, env_s_acc, env_v_acc, E);
  node_linear<<<dim3((Nn * 16 + 255) / 256), dim3(256), 0, stream>>>(
      env_s_acc, env_v_acc, envlin_ws, envlin_wv, env_s_lin, env_v_lin, Nn);
  hipMemsetAsync(envacc, 0, (size_t)Nn * 64 * 4, stream);
  edge_layer<0><<<dim3((E + 31) / 32), dim3(64), 0, stream>>>(
      edge_attr, eidx, lat, fs, fv, cut, env_s_lin, env_v_lin, env_s_acc, env_v_acc,
      wsw + woff[8], wsw + woff[9], wsw + woff[2], wsw + woff[3],
      wsw + woff[6], wsw + woff[7], wsw + woff[12], wsw + woff[13],
      (float*)d_out, E);
  node_linear<<<dim3((Nn * 16 + 255) / 256), dim3(256), 0, stream>>>(
      env_s_acc, env_v_acc, envlin_ws + 256, envlin_wv + 256, env_s_lin, env_v_lin, Nn);
  edge_layer<1><<<dim3((E + 31) / 32), dim3(64), 0, stream>>>(
      edge_attr, eidx, lat, fs, fv, cut, env_s_lin, env_v_lin, env_s_acc, env_v_acc,
      wsw + woff[10], wsw + woff[11], wsw + woff[2], wsw + woff[3],
      wsw + woff[6], wsw + woff[7], wsw + woff[12], wsw + woff[13],
      (float*)d_out, E);
}
